// SelectiveSSM_18305150615687
// MI455X (gfx1250) — matrix operand order repaired and checked by reading, not compiled, not run on hardware
//
#include <hip/hip_runtime.h>

#define Wd 256
#define Hd 256
#define Cc 4
#define Bb 2
#define MID 128
#define HW (Hd * Wd)                 // 65536 per (b,c) plane
#define NPIX (Bb * Cc * Hd * Wd)     // 524288
#define TW 20                        // conv1 LDS tile width: 16 + 2*dil (dil<=2)

typedef __attribute__((ext_vector_type(16))) _Float16 v16h;
typedef __attribute__((ext_vector_type(8)))  _Float16 v8h;
typedef __attribute__((ext_vector_type(8)))  float    v8f;

__device__ __forceinline__ float sigmoidf_(float x) {
  return 1.f / (1.f + __expf(-x));
}

// ---------------- weight packing (once per launch, tiny) --------------------
// w1t[n][64]: K = c*9+r padded 36->64 with zeros, f16.
__global__ void pack_w1(const float* __restrict__ w, _Float16* __restrict__ wt) {
  int i = blockIdx.x * 256 + threadIdx.x;     // 128*64 = 8192
  if (i >= MID * 64) return;
  int n = i >> 6, K = i & 63;
  wt[i] = (K < 36) ? (_Float16)w[n * 36 + K] : (_Float16)0.f;
}

// w2t[nSlot16][1152]: K' = r*128 + c (position-major), zeros for n>=4, f16.
__global__ void pack_w2(const float* __restrict__ w, _Float16* __restrict__ wt) {
  int i = blockIdx.x * 256 + threadIdx.x;     // 16*1152 = 18432
  if (i >= 16 * 1152) return;
  int n = i / 1152, kp = i - n * 1152;
  int r = kp >> 7, c = kp & 127;
  wt[i] = (n < Cc) ? (_Float16)w[n * 1152 + c * 9 + r] : (_Float16)0.f;
}

// ---------------- conv1: 4 -> 128, 3x3 (dil 1|2), ReLU, f16 channel-last ----
// grid (W/16, H, B), block 256 (8 waves). Wave w -> out channels [16w,16w+16).
// Input halo tile staged once in LDS; A fragments built from LDS with
// compile-time K decomposition (hi-half selected) -> no divides, no branches.
__global__ void conv1_relu_wmma(const float* __restrict__ st,
                                const _Float16* __restrict__ w1t,
                                const float* __restrict__ b1,
                                _Float16* __restrict__ tmid, int dil) {
  __shared__ _Float16 tile[Cc * 3 * TW];   // [c][row(3)][col(TW)]
  const int lane = threadIdx.x & 31;
  const int wave = threadIdx.x >> 5;
  const int hi   = lane >> 4;
  const int m    = lane & 15;
  const int x0   = blockIdx.x << 4;
  const int y    = blockIdx.y;
  const int b    = blockIdx.z;
  const int n    = (wave << 4) + (lane & 15);

  // cooperative halo fill: 240 elements, only bounds-checked loads here
  {
    int i = threadIdx.x;
    if (i < Cc * 3 * TW) {
      int ci  = i / (3 * TW);
      int rr  = (i / TW) % 3;
      int col = i % TW;
      int yy = y + (rr - 1) * dil;
      int xx = x0 - dil + col;
      float v = 0.f;
      if (yy >= 0 && yy < Hd && xx >= 0 && xx < Wd)
        v = st[((b * Cc + ci) * Hd + yy) * Wd + xx];
      tile[i] = (_Float16)v;
    }
  }
  __syncthreads();

  v8f c = {};
#pragma unroll
  for (int s = 0; s < 2; ++s) {       // K = 36 padded to 64: two k-steps
    v16h a, bm;
#pragma unroll
    for (int e = 0; e < 16; ++e) {
      // K for lane-half 0 / 1 are compile-time constants after unroll
      const int K0 = (s << 5) + (e < 8 ? e : e + 8);
      const int K1 = K0 + 8;
      const int c0 = K0 / 9, r0 = K0 - 9 * c0;
      const int c1 = K1 / 9, r1 = K1 - 9 * c1;
      const bool ok0 = (K0 < 36), ok1 = (K1 < 36);
      int idx0 = (c0 * 3 + r0 / 3) * TW + (r0 % 3) * dil + m;
      int idx1 = (c1 * 3 + r1 / 3) * TW + (r1 % 3) * dil + m;
      bool ok = hi ? ok1 : ok0;
      int idx = hi ? idx1 : idx0;
      _Float16 t = tile[ok ? idx : 0];   // LDS load always in-bounds
      a[e] = ok ? t : (_Float16)0.f;
    }
    const v8h* pb = (const v8h*)(w1t + n * 64 + (s << 5) + (hi << 3));
    v8h b0 = pb[0], b1v = pb[2];
#pragma unroll
    for (int e = 0; e < 8; ++e) { bm[e] = b0[e]; bm[e + 8] = b1v[e]; }
    c = __builtin_amdgcn_wmma_f32_16x16x32_f16(false, a, false, bm,
                                               (short)0, c, false, false);
  }
  const float bias = b1[n];
#pragma unroll
  for (int r = 0; r < 8; ++r) {
    int mm = r + (hi << 3);             // D: M = r + 8*hiHalf, N = lane&15
    float v = c[r] + bias;
    v = v > 0.f ? v : 0.f;
    tmid[((((b * Hd + y) * Wd) + x0 + mm) << 7) + n] = (_Float16)v;
  }
}

// ---------------- conv2: 128 -> 4, 3x3 (dil 1|2), accumulate into acc -------
// grid (W/128, H, B), block 256 (8 waves); wave w -> strip x0 = (8*bx+w)*16.
// K' = 1152 position-major -> 9 taps x 4 k-steps of 32 channels.
// A fragment per k-step = two contiguous b128 loads (channel-last tmid);
// B fragment = two contiguous b128 loads from packed w2t. No divergence.
__global__ void conv2_acc_wmma(const _Float16* __restrict__ tmid,
                               const _Float16* __restrict__ w2t,
                               const float* __restrict__ b2,
                               float* __restrict__ acc, int dil, int add) {
  const int lane = threadIdx.x & 31;
  const int wave = threadIdx.x >> 5;
  const int hi   = lane >> 4;
  const int m    = lane & 15;
  const int x0   = ((blockIdx.x << 3) + wave) << 4;
  const int y    = blockIdx.y;
  const int b    = blockIdx.z;
  const int n    = lane & 15;

  __builtin_prefetch(w2t + n * 1152, 0, 1);
  const v8h z8 = {};
  v8f c = {};
#pragma unroll
  for (int r9 = 0; r9 < 9; ++r9) {       // 3x3 tap
    const int ky = r9 / 3 - 1, kx = r9 % 3 - 1;   // constants after unroll
    int yy = y + ky * dil;
    int xx = x0 + m + kx * dil;
    bool inb = (yy >= 0) & (yy < Hd) & (xx >= 0) & (xx < Wd);
    int yyc = min(max(yy, 0), Hd - 1);
    int xxc = min(max(xx, 0), Wd - 1);
    const _Float16* pix = tmid + ((((b * Hd + yyc) * Wd) + xxc) << 7);
#pragma unroll
    for (int q = 0; q < 4; ++q) {        // 32-channel chunks
      const v8h* pa = (const v8h*)(pix + (q << 5) + (hi << 3));
      v8h alo = inb ? pa[0] : z8;        // channels c0..c0+7
      v8h ahi = inb ? pa[2] : z8;        // channels c0+16..c0+23
      v16h a, bm;
#pragma unroll
      for (int e = 0; e < 8; ++e) { a[e] = alo[e]; a[e + 8] = ahi[e]; }
      const v8h* pb = (const v8h*)(w2t + n * 1152 + (r9 << 7) + (q << 5) + (hi << 3));
      v8h b0 = pb[0], b1v = pb[2];
#pragma unroll
      for (int e = 0; e < 8; ++e) { bm[e] = b0[e]; bm[e + 8] = b1v[e]; }
      c = __builtin_amdgcn_wmma_f32_16x16x32_f16(false, a, false, bm,
                                                 (short)0, c, false, false);
    }
  }
  if (n < Cc) {
    const float bias = b2[n];
#pragma unroll
    for (int r = 0; r < 8; ++r) {
      int mm = r + (hi << 3);
      int idx = ((b * Cc + n) * Hd + y) * Wd + x0 + mm;
      float v = c[r] + bias;
      acc[idx] = add ? (acc[idx] + v) : v;
    }
  }
}

// ---------------- depthwise sobel -> gradient magnitude ---------------------
__global__ void sobel_gm(const float* __restrict__ st, float* __restrict__ gm) {
  int idx = blockIdx.x * 256 + threadIdx.x;
  if (idx >= NPIX) return;
  int x = idx & 255, y = (idx >> 8) & 255, bc = idx >> 16;
  const float* p = st + (bc << 16);
  auto L = [&](int yy, int xx) -> float {
    if (yy < 0 || yy >= Hd || xx < 0 || xx >= Wd) return 0.f;
    return p[(yy << 8) + xx];
  };
  float a00 = L(y - 1, x - 1), a01 = L(y - 1, x), a02 = L(y - 1, x + 1);
  float a10 = L(y, x - 1), a12 = L(y, x + 1);
  float a20 = L(y + 1, x - 1), a21 = L(y + 1, x), a22 = L(y + 1, x + 1);
  float gx = -a00 + a02 - 2.f * a10 + 2.f * a12 - a20 + a22;
  float gy = -a00 - 2.f * a01 - a02 + a20 + 2.f * a21 + a22;
  gm[idx] = sqrtf(gx * gx + gy * gy);
}

// ---------------- 5x5 box sums + unnormalized sim + per-(b,c) sum -----------
__global__ void boxsim(const float* __restrict__ gm, const float* __restrict__ st,
                       float* __restrict__ simun, float* __restrict__ bsst,
                       float* __restrict__ simsum) {
  __shared__ float sd[256];
  int bc = blockIdx.y;
  int p  = (blockIdx.x << 8) + threadIdx.x;
  int y = p >> 8, x = p & 255;
  const float* g = gm + (bc << 16);
  const float* s = st + (bc << 16);
  float sg = 0.f, sg2 = 0.f, ss = 0.f;
#pragma unroll
  for (int dy = -2; dy <= 2; ++dy) {
    int yy = y + dy;
    if (yy < 0 || yy >= Hd) continue;
#pragma unroll
    for (int dx = -2; dx <= 2; ++dx) {
      int xx = x + dx;
      if (xx < 0 || xx >= Wd) continue;
      float gv = g[(yy << 8) + xx];
      sg += gv; sg2 += gv * gv;
      ss += s[(yy << 8) + xx];
    }
  }
  float gc = g[p];
  float dist = sg2 - 2.f * gc * sg + 25.f * gc * gc;
  float sim = __expf(-dist * 2.0f);   // 1/(2*sigma^2) = 2, sigma = 0.5
  simun[(bc << 16) + p] = sim;
  bsst[(bc << 16) + p]  = ss;
  sd[threadIdx.x] = sim;
  __syncthreads();
  for (int off = 128; off > 0; off >>= 1) {
    if (threadIdx.x < off) sd[threadIdx.x] += sd[threadIdx.x + off];
    __syncthreads();
  }
  if (threadIdx.x == 0) atomicAdd(&simsum[bc], sd[0]);
}

// ---------------- combine: weighted + acc + dynamic 4->4 conv ---------------
__global__ void combine(const float* __restrict__ cur, const float* __restrict__ simun,
                        const float* __restrict__ bsst, const float* __restrict__ simsum,
                        const float* __restrict__ acc, const float* __restrict__ dcw,
                        const float* __restrict__ dcb, float* __restrict__ nxt) {
  int idx = blockIdx.x * 256 + threadIdx.x;
  if (idx >= NPIX) return;
  int x = idx & 255, y = (idx >> 8) & 255, bc = idx >> 16;
  int b = bc >> 2, c = bc & 3;
  float hf = dcb[c];
#pragma unroll
  for (int c2 = 0; c2 < Cc; ++c2) {
    const float* p = cur + (((b * Cc + c2)) << 16);
#pragma unroll
    for (int ky = -1; ky <= 1; ++ky) {
      int yy = y + ky;
      if (yy < 0 || yy >= Hd) continue;
#pragma unroll
      for (int kx = -1; kx <= 1; ++kx) {
        int xx = x + kx;
        if (xx < 0 || xx >= Wd) continue;
        hf += dcw[((c * Cc + c2) * 3 + (ky + 1)) * 3 + (kx + 1)] * p[(yy << 8) + xx];
      }
    }
  }
  float sim = simun[idx] / (simsum[bc] + 1e-8f);
  nxt[idx] = sim * bsst[idx] + acc[idx] + hf;
}

// ---------------- finalize: branch select, sel, out, loss -------------------
__global__ void finalize(const float* __restrict__ st, const float* __restrict__ dcw,
                         const float* __restrict__ dcb, const int* __restrict__ tmod,
                         float* __restrict__ out, float* __restrict__ loss) {
  __shared__ float sd[256];
  int idx = blockIdx.x * 256 + threadIdx.x;
  int x = idx & 255, y = (idx >> 8) & 255, bc = idx >> 16;
  int b = bc >> 2, c = bc & 3;
  const float* p = st + (bc << 16);
  auto L = [&](int yy, int xx) -> float {
    if (yy < 0 || yy >= Hd || xx < 0 || xx >= Wd) return 0.f;
    return p[(yy << 8) + xx];
  };
  float a00 = L(y - 1, x - 1), a01 = L(y - 1, x), a02 = L(y - 1, x + 1);
  float a10 = L(y, x - 1), a12 = L(y, x + 1);
  float a20 = L(y + 1, x - 1), a21 = L(y + 1, x), a22 = L(y + 1, x + 1);
  float gx = -a00 + a02 - 2.f * a10 + 2.f * a12 - a20 + a22;
  float gy = -a00 - 2.f * a01 - a02 + a20 + 2.f * a21 + a22;
  float gmv = sqrtf(gx * gx + gy * gy);

  float hfl = dcb[c];
#pragma unroll
  for (int c2 = 0; c2 < Cc; ++c2) {
    const float* q = st + (((b * Cc + c2)) << 16);
#pragma unroll
    for (int ky = -1; ky <= 1; ++ky) {
      int yy = y + ky;
      if (yy < 0 || yy >= Hd) continue;
#pragma unroll
      for (int kx = -1; kx <= 1; ++kx) {
        int xx = x + kx;
        if (xx < 0 || xx >= Wd) continue;
        hfl += dcw[((c * Cc + c2) * 3 + (ky + 1)) * 3 + (kx + 1)] * q[(yy << 8) + xx];
      }
    }
  }
  float hf  = sigmoidf_(hfl);
  float sgm = sigmoidf_(gmv);
  float gh  = sigmoidf_(gmv * hf);
  int t = tmod[b];
  float pos, neg;
  if (t == 0)      { pos = hf;       neg = 1.f - sgm; }
  else if (t == 1) { pos = sgm;      neg = 1.f - hf; }
  else if (t == 2) { pos = hf * sgm; neg = 1.f - hf * sgm; }
  else             { pos = gh;       neg = 1.f - gh; }
  pos = fminf(fmaxf(pos, 0.f), 1.f);
  neg = fminf(fmaxf(neg, 0.f), 1.f);
  float ep = __expf(pos), en = __expf(neg);
  float sel = ep / (ep + en + 1e-8f);
  out[idx] = p[(y << 8) + x] * sel;

  sd[threadIdx.x] = sel * neg + (1.f - sel) * pos;
  __syncthreads();
  for (int off = 128; off > 0; off >>= 1) {
    if (threadIdx.x < off) sd[threadIdx.x] += sd[threadIdx.x + off];
    __syncthreads();
  }
  if (threadIdx.x == 0) atomicAdd(loss, sd[0] * (1.0f / (float)NPIX));
}

// ---------------- gating ----------------------------------------------------
__global__ void reduce_mean(const float* __restrict__ x, float* __restrict__ gfeat) {
  __shared__ float sd[256];
  int bc = blockIdx.x;
  const float* p = x + (bc << 16);
  float s = 0.f;
  for (int i = threadIdx.x; i < HW; i += 256) s += p[i];
  sd[threadIdx.x] = s;
  __syncthreads();
  for (int off = 128; off > 0; off >>= 1) {
    if (threadIdx.x < off) sd[threadIdx.x] += sd[threadIdx.x + off];
    __syncthreads();
  }
  if (threadIdx.x == 0) gfeat[bc] = sd[0] * (1.0f / (float)HW);
}

__global__ void gate_small(const float* __restrict__ gfeat, const float* __restrict__ fcw,
                           const float* __restrict__ fcb, const float* __restrict__ mw,
                           float* __restrict__ gate) {
  int t = threadIdx.x;
  if (t >= Bb * Cc) return;
  int b = t >> 2, c = t & 3;
  float z = fcb[c];
  for (int j = 0; j < Cc; ++j) z += fcw[c * Cc + j] * gfeat[b * Cc + j];
  gate[t] = sigmoidf_(z) * sigmoidf_(mw[c]);
}

__global__ void init_state(const float* __restrict__ x, const float* __restrict__ gate,
                           float* __restrict__ st) {
  int idx = blockIdx.x * 256 + threadIdx.x;
  if (idx >= NPIX) return;
  st[idx] = x[idx] * gate[idx >> 16];
}

__global__ void zero_small(float* __restrict__ p, int n) {
  int i = threadIdx.x;
  if (i < n) p[i] = 0.f;
}

extern "C" void kernel_launch(void* const* d_in, const int* in_sizes, int n_in,
                              void* d_out, int out_size, void* d_ws, size_t ws_size,
                              hipStream_t stream) {
  (void)in_sizes; (void)n_in; (void)out_size; (void)ws_size;
  const float* x     = (const float*)d_in[0];
  const int*   tmod  = (const int*)d_in[1];
  const float* fcw   = (const float*)d_in[2];
  const float* fcb   = (const float*)d_in[3];
  const float* mw    = (const float*)d_in[4];
  const float* su_w1 = (const float*)d_in[5];
  const float* su_b1 = (const float*)d_in[6];
  const float* su_w2 = (const float*)d_in[7];
  const float* su_b2 = (const float*)d_in[8];
  const float* nu_w1 = (const float*)d_in[9];
  const float* nu_b1 = (const float*)d_in[10];
  const float* nu_w2 = (const float*)d_in[11];
  const float* nu_b2 = (const float*)d_in[12];
  const float* dcw   = (const float*)d_in[13];
  const float* dcb   = (const float*)d_in[14];
  // d_in[15], d_in[16]: sobel kernels (constant, hardcoded in-kernel)

  float* ws     = (float*)d_ws;
  float* stateA = ws;
  float* stateB = stateA + NPIX;
  float* accb   = stateB + NPIX;
  float* gmb    = accb + NPIX;
  float* simun  = gmb + NPIX;
  float* bsst   = simun + NPIX;
  float* gfeat  = bsst + NPIX;
  float* gate   = gfeat + 8;
  float* simsum = gate + 8;
  _Float16* tmid   = (_Float16*)(simsum + 8);       // [B][H][W][128] f16
  _Float16* w1t_su = tmid + (size_t)Bb * Hd * Wd * MID;
  _Float16* w1t_nu = w1t_su + MID * 64;
  _Float16* w2t_su = w1t_nu + MID * 64;
  _Float16* w2t_nu = w2t_su + 16 * 1152;

  float* out  = (float*)d_out;
  float* loss = out + NPIX;

  zero_small<<<1, 32, 0, stream>>>(loss, 1);
  pack_w1<<<(MID * 64 + 255) / 256, 256, 0, stream>>>(su_w1, w1t_su);
  pack_w1<<<(MID * 64 + 255) / 256, 256, 0, stream>>>(nu_w1, w1t_nu);
  pack_w2<<<(16 * 1152 + 255) / 256, 256, 0, stream>>>(su_w2, w2t_su);
  pack_w2<<<(16 * 1152 + 255) / 256, 256, 0, stream>>>(nu_w2, w2t_nu);
  reduce_mean<<<Bb * Cc, 256, 0, stream>>>(x, gfeat);
  gate_small<<<1, 32, 0, stream>>>(gfeat, fcw, fcb, mw, gate);
  init_state<<<NPIX / 256, 256, 0, stream>>>(x, gate, stateA);

  float* cur = stateA;
  float* nxt = stateB;
  for (int it = 0; it < 3; ++it) {
    conv1_relu_wmma<<<dim3(Wd / 16, Hd, Bb), 256, 0, stream>>>(cur, w1t_su, su_b1, tmid, 1);
    conv2_acc_wmma <<<dim3(Wd / 128, Hd, Bb), 256, 0, stream>>>(tmid, w2t_su, su_b2, accb, 1, 0);
    conv1_relu_wmma<<<dim3(Wd / 16, Hd, Bb), 256, 0, stream>>>(cur, w1t_nu, nu_b1, tmid, 2);
    conv2_acc_wmma <<<dim3(Wd / 128, Hd, Bb), 256, 0, stream>>>(tmid, w2t_nu, nu_b2, accb, 2, 1);
    sobel_gm<<<NPIX / 256, 256, 0, stream>>>(cur, gmb);
    zero_small<<<1, 32, 0, stream>>>(simsum, 8);
    boxsim<<<dim3(HW / 256, Bb * Cc), 256, 0, stream>>>(gmb, cur, simun, bsst, simsum);
    combine<<<NPIX / 256, 256, 0, stream>>>(cur, simun, bsst, simsum, accb, dcw, dcb, nxt);
    float* t = cur; cur = nxt; nxt = t;
  }
  finalize<<<NPIX / 256, 256, 0, stream>>>(cur, dcw, dcb, tmod, out, loss);
}
